// MultiHeadAttentionLayer_29420525978329
// MI455X (gfx1250) — compile-verified, matrix operand order checked by reading, not run on hardware
//
#include <hip/hip_runtime.h>

// ---------------- problem constants ----------------
#define BB   2
#define SS   2048
#define DD   1024
#define HH   16
#define DKV  64
#define TOK  (BB*SS)      // 4096 tokens
#define LN_EPS 1e-5f

// ---------------- CDNA5 WMMA types ----------------
typedef __bf16 bf16_t;
typedef __attribute__((ext_vector_type(16))) __bf16 v16bf;
typedef __attribute__((ext_vector_type(8)))  __bf16 v8bf;
typedef __attribute__((ext_vector_type(8)))  float  v8f;

union AFrag { v16bf v; v8bf h[2]; };

__device__ __forceinline__ v8f wmma_bf16(v16bf a, v16bf b, v8f c) {
  // v_wmma_f32_16x16x32_bf16: D = A(16x32) * B(32x16) + C(16x16 f32)
  return __builtin_amdgcn_wmma_f32_16x16x32_bf16(
      /*neg_a=*/false, a, /*neg_b=*/false, b,
      /*c_mod=*/(short)0, c, /*reuse_a=*/false, /*reuse_b=*/false);
}

__device__ __forceinline__ v8f zero8() {
  v8f z = {0.f,0.f,0.f,0.f,0.f,0.f,0.f,0.f};
  return z;
}

// ======================================================================
// Kernel 0: fp32 -> bf16 convert for X; transpose+convert all weights so
// both WMMA operands are K-major (each lane loads two contiguous 16-byte
// runs matching the 16-bit A/B fragment striping).
// ======================================================================
__global__ __launch_bounds__(256) void convert_kernel(
    const float* __restrict__ X,
    const float* __restrict__ Wq, const float* __restrict__ Wk,
    const float* __restrict__ Wv, const float* __restrict__ Wo,
    bf16_t* __restrict__ Xb,
    bf16_t* __restrict__ Wqt, bf16_t* __restrict__ Wkt,
    bf16_t* __restrict__ Wvt, bf16_t* __restrict__ Wot)
{
  int tid = blockIdx.x * blockDim.x + threadIdx.x;
  if (tid < TOK*DD) Xb[tid] = (bf16_t)X[tid];
  if (tid < DD*DD) {
    int k = tid >> 10;          // input row  (K dim)
    int n = tid & (DD-1);       // input col  (N dim)
    int tpos = n*DD + k;        // transposed: [N][K]
    Wqt[tpos] = (bf16_t)Wq[tid];
    Wkt[tpos] = (bf16_t)Wk[tid];
    Wvt[tpos] = (bf16_t)Wv[tid];
    Wot[tpos] = (bf16_t)Wo[tid];
  }
}

// ----------------------------------------------------------------------
// Shared GEMM mainloop: each wave computes a 32(M) x 64(N) strip with
// double-buffered fragments so loads for step i+1 are issued before the
// WMMAs of step i (hides VMEM latency behind the matrix pipe).
// acc[r][t]: r = M half-tile (0: rows mbase..+15, 1: rows mbase+16..+31),
//            t = N tile of 16.
// ----------------------------------------------------------------------
struct GemmCtx {
  const bf16_t* arow0;
  const bf16_t* arow1;
  const bf16_t* brow[4];
  int half;
};

__device__ __forceinline__ void gemm_load_frags(
    const GemmCtx& g, int k0, AFrag (&Af)[2], AFrag (&Bf)[4])
{
  const int h8 = g.half*8;
  Af[0].h[0] = *(const v8bf*)(g.arow0 + k0 +      h8);
  Af[0].h[1] = *(const v8bf*)(g.arow0 + k0 + 16 + h8);
  Af[1].h[0] = *(const v8bf*)(g.arow1 + k0 +      h8);
  Af[1].h[1] = *(const v8bf*)(g.arow1 + k0 + 16 + h8);
  #pragma unroll
  for (int t=0;t<4;++t) {
    Bf[t].h[0] = *(const v8bf*)(g.brow[t] + k0 +      h8);
    Bf[t].h[1] = *(const v8bf*)(g.brow[t] + k0 + 16 + h8);
  }
}

__device__ __forceinline__ void gemm_mma(
    AFrag (&Af)[2], AFrag (&Bf)[4], v8f (&acc)[2][4])
{
  #pragma unroll
  for (int r=0;r<2;++r)
    #pragma unroll
    for (int t=0;t<4;++t)
      acc[r][t] = wmma_bf16(Af[r].v, Bf[t].v, acc[r][t]);
}

__device__ __forceinline__ void gemm_mainloop(const GemmCtx& g, v8f (&acc)[2][4])
{
  AFrag Af0[2], Bf0[4], Af1[2], Bf1[4];
  gemm_load_frags(g, 0, Af0, Bf0);
  // 2x hand-unrolled software pipeline (DD/64 = 16 outer steps)
  for (int k0 = 0; k0 < DD; k0 += 64) {
    __builtin_prefetch(g.arow0 + k0 + 256, 0, 1);
    gemm_load_frags(g, k0 + 32, Af1, Bf1);     // issue next loads first
    gemm_mma(Af0, Bf0, acc);                   // consume current buffer
    if (k0 + 64 < DD) gemm_load_frags(g, k0 + 64, Af0, Bf0);
    gemm_mma(Af1, Bf1, acc);
  }
}

// ======================================================================
// Kernel 1: fused QKV projection GEMM (bf16 WMMA, f32 accum).
// Block = 128 threads (4 waves). Each wave computes 32(M) x 64(N).
// blockIdx.z selects Q/K/V. V is stored transposed: Vt[b][h][dv][S].
// ======================================================================
__global__ __launch_bounds__(128) void qkv_gemm_kernel(
    const bf16_t* __restrict__ Xb,
    const bf16_t* __restrict__ Wqt, const bf16_t* __restrict__ Wkt,
    const bf16_t* __restrict__ Wvt,
    const float* __restrict__ bq, const float* __restrict__ bk,
    const float* __restrict__ bv,
    bf16_t* __restrict__ Q, bf16_t* __restrict__ K, bf16_t* __restrict__ Vt)
{
  const int which = blockIdx.z;
  const bf16_t* Wt   = (which==0) ? Wqt : (which==1) ? Wkt : Wvt;
  const float*  bias = (which==0) ? bq  : (which==1) ? bk  : bv;

  const int lane = threadIdx.x & 31;
  const int wave = threadIdx.x >> 5;
  const int ln   = lane & 15;
  const int half = lane >> 4;
  const int mbase = blockIdx.x*128 + wave*32;
  const int nbase = blockIdx.y*64;

  GemmCtx g;
  g.half  = half;
  g.arow0 = Xb + (size_t)(mbase +      ln)*DD;
  g.arow1 = Xb + (size_t)(mbase + 16 + ln)*DD;
  #pragma unroll
  for (int t=0;t<4;++t) g.brow[t] = Wt + (size_t)(nbase + t*16 + ln)*DD;

  v8f acc[2][4];
  #pragma unroll
  for (int r=0;r<2;++r)
    #pragma unroll
    for (int t=0;t<4;++t) acc[r][t] = zero8();

  gemm_mainloop(g, acc);

  // C layout: VGPR v, lanes 0-15 -> M=v ; lanes 16-31 -> M=v+8 ; N = ln
  #pragma unroll
  for (int r=0;r<2;++r) {
    #pragma unroll
    for (int t=0;t<4;++t) {
      #pragma unroll
      for (int v=0; v<8; ++v) {
        int row = mbase + r*16 + v + 8*half;
        int col = nbase + t*16 + ln;
        float val = acc[r][t][v] + bias[col];
        if (which == 0) {
          Q[(size_t)row*DD + col] = (bf16_t)val;
        } else if (which == 1) {
          K[(size_t)row*DD + col] = (bf16_t)val;
        } else {
          int bb2 = row >> 11, s = row & (SS-1);
          int hh  = col >> 6,  d = col & 63;
          Vt[(((size_t)(bb2*HH + hh))*DKV + d)*SS + s] = (bf16_t)val;
        }
      }
    }
  }
}

// ======================================================================
// Kernel 2: flash attention. Block = 128 threads (4 waves), each block
// handles one (b,h) and 64 query rows; each wave owns 16 query rows.
// Online softmax over 32-key blocks: 4 score WMMAs + 4 PV WMMAs each.
// V fragments are loaded right after the score WMMAs are issued so their
// global latency overlaps the softmax VALU work. P is re-laid-out C->A
// through a per-wave LDS tile (intra-wave DS is in-order on CDNA5).
// ======================================================================
__global__ __launch_bounds__(128) void flash_attn_kernel(
    const bf16_t* __restrict__ Q, const bf16_t* __restrict__ K,
    const bf16_t* __restrict__ Vt, bf16_t* __restrict__ AO)
{
  const int lane = threadIdx.x & 31;
  const int wave = threadIdx.x >> 5;
  const int ln   = lane & 15;
  const int half = lane >> 4;
  const int bh = blockIdx.y;
  const int b  = bh >> 4, h = bh & 15;
  const int t0 = b*SS + blockIdx.x*64 + wave*16;   // wave's q-row base token

  __shared__ __align__(16) bf16_t Plds[4][16][32];

  // Q fragments (dk=64 -> two K=32 fragments), resident for whole loop
  const bf16_t* qrow = Q + (size_t)(t0 + ln)*DD + h*DKV;
  AFrag q0, q1;
  q0.h[0] = *(const v8bf*)(qrow +      half*8);
  q0.h[1] = *(const v8bf*)(qrow + 16 + half*8);
  q1.h[0] = *(const v8bf*)(qrow + 32 + half*8);
  q1.h[1] = *(const v8bf*)(qrow + 48 + half*8);

  v8f o[4];
  #pragma unroll
  for (int t=0;t<4;++t) o[t] = zero8();
  float rm[8], rl[8];
  #pragma unroll
  for (int v=0;v<8;++v) { rm[v] = -1e30f; rl[v] = 0.f; }

  const bf16_t* kbasep = K  + (size_t)(b*SS)*DD + h*DKV;
  const bf16_t* vbase  = Vt + (size_t)(b*HH + h)*DKV*SS;

  for (int kb = 0; kb < SS; kb += 32) {
    // ---- scores: S = Q * K^T (two 16-key N tiles) ----
    const bf16_t* kr0 = kbasep + (size_t)(kb      + ln)*DD;
    const bf16_t* kr1 = kbasep + (size_t)(kb + 16 + ln)*DD;
    AFrag k00,k01,k10,k11;
    k00.h[0] = *(const v8bf*)(kr0 +      half*8);
    k00.h[1] = *(const v8bf*)(kr0 + 16 + half*8);
    k01.h[0] = *(const v8bf*)(kr0 + 32 + half*8);
    k01.h[1] = *(const v8bf*)(kr0 + 48 + half*8);
    k10.h[0] = *(const v8bf*)(kr1 +      half*8);
    k10.h[1] = *(const v8bf*)(kr1 + 16 + half*8);
    k11.h[0] = *(const v8bf*)(kr1 + 32 + half*8);
    k11.h[1] = *(const v8bf*)(kr1 + 48 + half*8);

    v8f s0 = wmma_bf16(q0.v, k00.v, zero8());
    s0     = wmma_bf16(q1.v, k01.v, s0);
    v8f s1 = wmma_bf16(q0.v, k10.v, zero8());
    s1     = wmma_bf16(q1.v, k11.v, s1);

    // ---- issue V fragment loads now; latency hides under softmax ----
    AFrag vb[4];
    #pragma unroll
    for (int t=0;t<4;++t) {
      const bf16_t* vr = vbase + (size_t)(t*16 + ln)*SS + kb;
      vb[t].h[0] = *(const v8bf*)(vr +      half*8);
      vb[t].h[1] = *(const v8bf*)(vr + 16 + half*8);
    }

    // ---- online softmax (per-row stats via 16-lane xor reductions) ----
    #pragma unroll
    for (int v=0; v<8; ++v) {
      float x0 = s0[v]*0.125f, x1 = s1[v]*0.125f;   // 1/sqrt(64)
      float mx = fmaxf(x0, x1);
      #pragma unroll
      for (int off=1; off<16; off<<=1) mx = fmaxf(mx, __shfl_xor(mx, off, 32));
      float mnew = fmaxf(rm[v], mx);
      float corr = __expf(rm[v] - mnew);
      float p0 = __expf(x0 - mnew), p1 = __expf(x1 - mnew);
      float sump = p0 + p1;
      #pragma unroll
      for (int off=1; off<16; off<<=1) sump += __shfl_xor(sump, off, 32);
      rl[v] = rl[v]*corr + sump;
      rm[v] = mnew;
      o[0][v] *= corr; o[1][v] *= corr; o[2][v] *= corr; o[3][v] *= corr;
      // scatter P into the per-wave LDS tile in row-major [16][32]
      Plds[wave][8*half + v][ln]      = (bf16_t)p0;
      Plds[wave][8*half + v][16 + ln] = (bf16_t)p1;
    }

    // ---- re-load P as an A fragment (16x32) ----
    AFrag pa;
    pa.h[0] = *(const v8bf*)&Plds[wave][ln][     half*8];
    pa.h[1] = *(const v8bf*)&Plds[wave][ln][16 + half*8];

    // ---- O += P * V  (V^T rows are dv, cols are key tokens) ----
    #pragma unroll
    for (int t=0;t<4;++t) o[t] = wmma_bf16(pa.v, vb[t].v, o[t]);
  }

  // ---- normalize and emit bf16 attention output [token][H*dv] ----
  #pragma unroll
  for (int t=0;t<4;++t) {
    #pragma unroll
    for (int v=0; v<8; ++v) {
      int tok = t0 + v + 8*half;
      float val = o[t][v] / rl[v];
      AO[(size_t)tok*DD + h*DKV + t*16 + ln] = (bf16_t)val;
    }
  }
}

// ======================================================================
// Kernel 3: out projection GEMM + bias + residual (f32 out).
// ======================================================================
__global__ __launch_bounds__(128) void oproj_kernel(
    const bf16_t* __restrict__ AO, const bf16_t* __restrict__ Wot,
    const float* __restrict__ bo, const float* __restrict__ resid,
    float* __restrict__ res)
{
  const int lane = threadIdx.x & 31;
  const int wave = threadIdx.x >> 5;
  const int ln   = lane & 15;
  const int half = lane >> 4;
  const int mbase = blockIdx.x*128 + wave*32;
  const int nbase = blockIdx.y*64;

  GemmCtx g;
  g.half  = half;
  g.arow0 = AO + (size_t)(mbase +      ln)*DD;
  g.arow1 = AO + (size_t)(mbase + 16 + ln)*DD;
  #pragma unroll
  for (int t=0;t<4;++t) g.brow[t] = Wot + (size_t)(nbase + t*16 + ln)*DD;

  v8f acc[2][4];
  #pragma unroll
  for (int r=0;r<2;++r)
    #pragma unroll
    for (int t=0;t<4;++t) acc[r][t] = zero8();

  gemm_mainloop(g, acc);

  #pragma unroll
  for (int r=0;r<2;++r) {
    #pragma unroll
    for (int t=0;t<4;++t) {
      #pragma unroll
      for (int v=0; v<8; ++v) {
        int row = mbase + r*16 + v + 8*half;
        int col = nbase + t*16 + ln;
        res[(size_t)row*DD + col] =
            acc[r][t][v] + bo[col] + resid[(size_t)row*DD + col];
      }
    }
  }
}

// ======================================================================
// Kernel 4: LayerNorm over D=1024 per token.
// ======================================================================
__global__ __launch_bounds__(256) void ln_kernel(
    const float* __restrict__ res, const float* __restrict__ gamma,
    const float* __restrict__ beta, float* __restrict__ out)
{
  __shared__ float ssum[256];
  __shared__ float ssq[256];
  const int row = blockIdx.x;
  const float* r = res + (size_t)row*DD;
  float s = 0.f, q = 0.f;
  for (int j = threadIdx.x; j < DD; j += 256) {
    float x = r[j];
    s += x; q += x*x;
  }
  ssum[threadIdx.x] = s; ssq[threadIdx.x] = q;
  __syncthreads();
  for (int st = 128; st > 0; st >>= 1) {
    if (threadIdx.x < st) {
      ssum[threadIdx.x] += ssum[threadIdx.x + st];
      ssq[threadIdx.x]  += ssq[threadIdx.x + st];
    }
    __syncthreads();
  }
  const float mu   = ssum[0] * (1.f/DD);
  const float var  = ssq[0] * (1.f/DD) - mu*mu;
  const float rinv = rsqrtf(var + LN_EPS);
  for (int j = threadIdx.x; j < DD; j += 256) {
    out[(size_t)row*DD + j] = (r[j] - mu)*rinv*gamma[j] + beta[j];
  }
}

// ======================================================================
// launch
// ======================================================================
extern "C" void kernel_launch(void* const* d_in, const int* in_sizes, int n_in,
                              void* d_out, int out_size, void* d_ws, size_t ws_size,
                              hipStream_t stream) {
  const float* queries = (const float*)d_in[0];
  const float* Wq = (const float*)d_in[1];
  const float* bq = (const float*)d_in[2];
  const float* Wk = (const float*)d_in[3];
  const float* bk = (const float*)d_in[4];
  const float* Wv = (const float*)d_in[5];
  const float* bv = (const float*)d_in[6];
  const float* Wo = (const float*)d_in[7];
  const float* bo = (const float*)d_in[8];
  const float* gamma = (const float*)d_in[9];
  const float* beta  = (const float*)d_in[10];

  char* ws = (char*)d_ws;
  size_t off = 0;
  auto alloc = [&](size_t bytes) -> void* {
    void* p = ws + off;
    off += (bytes + 255) & ~(size_t)255;
    return p;
  };
  bf16_t* Xb   = (bf16_t*)alloc((size_t)TOK*DD*sizeof(bf16_t));
  bf16_t* Wqt  = (bf16_t*)alloc((size_t)DD*DD*sizeof(bf16_t));
  bf16_t* Wkt  = (bf16_t*)alloc((size_t)DD*DD*sizeof(bf16_t));
  bf16_t* Wvt  = (bf16_t*)alloc((size_t)DD*DD*sizeof(bf16_t));
  bf16_t* Wot  = (bf16_t*)alloc((size_t)DD*DD*sizeof(bf16_t));
  bf16_t* Qb   = (bf16_t*)alloc((size_t)TOK*DD*sizeof(bf16_t));
  bf16_t* Kb   = (bf16_t*)alloc((size_t)TOK*DD*sizeof(bf16_t));
  bf16_t* Vtb  = (bf16_t*)alloc((size_t)TOK*DD*sizeof(bf16_t));
  bf16_t* AO   = (bf16_t*)alloc((size_t)TOK*DD*sizeof(bf16_t));
  float*  resb = (float*) alloc((size_t)TOK*DD*sizeof(float));
  (void)ws_size; (void)in_sizes; (void)n_in; (void)out_size;

  // 0) convert / transpose
  convert_kernel<<<(TOK*DD + 255)/256, 256, 0, stream>>>(
      queries, Wq, Wk, Wv, Wo, Xb, Wqt, Wkt, Wvt, Wot);

  // 1) fused QKV projection (z = which of Q/K/V)
  qkv_gemm_kernel<<<dim3(TOK/128, DD/64, 3), 128, 0, stream>>>(
      Xb, Wqt, Wkt, Wvt, bq, bk, bv, Qb, Kb, Vtb);

  // 2) flash attention
  flash_attn_kernel<<<dim3(SS/64, BB*HH), 128, 0, stream>>>(Qb, Kb, Vtb, AO);

  // 3) out projection + bias + residual
  oproj_kernel<<<dim3(TOK/128, DD/64), 128, 0, stream>>>(
      AO, Wot, bo, queries, resb);

  // 4) layernorm -> final output
  ln_kernel<<<TOK, 256, 0, stream>>>(resb, gamma, beta, (float*)d_out);
}
